// SAMNetPP_87273735455103
// MI455X (gfx1250) — matrix operand order repaired and checked by reading, not compiled, not run on hardware
//
#include <hip/hip_runtime.h>
#include <hip/hip_bf16.h>
#include <math.h>

// PointNet++ part-segmentation forward for gfx1250 (MI455X).
// GEMMs: pre-packed f16 operands in WMMA fragment layout, 16x64 tile per wave,
//        4x v_wmma_f32_16x16x32_f16 per K-chunk, f32 accumulate.
// FPS: one workgroup per batch, cloud + min-dist resident in LDS (wave32 reductions).
// Ball query: one wave32 per centroid, __ballot ordered compaction.
// Requires ws_size >= ~335 MB.

typedef __attribute__((ext_vector_type(16))) _Float16 v16h;
typedef __attribute__((ext_vector_type(8)))  _Float16 v8h;
typedef __attribute__((ext_vector_type(8)))  float    v8f;

#define BATCH 16
#define NPTS  4096
#define NCLS  50

// ---------------------------------------------------------------- input split
__global__ void k_split_input(const float* __restrict__ in,
                              float* __restrict__ oxyz, float* __restrict__ opts, int n) {
  int tid = blockIdx.x * blockDim.x + threadIdx.x;
  int total = BATCH * n * 6;
  if (tid >= total) return;
  int c = tid % 6; int nn = (tid / 6) % n; int b = tid / (6 * n);
  float v = in[((size_t)b * 6 + c) * n + nn];
  if (c < 3) oxyz[((size_t)b * n + nn) * 3 + c] = v;
  else       opts[((size_t)b * n + nn) * 3 + (c - 3)] = v;
}

// ---------------------------------------------------------------- FPS (1 WG / batch, LDS-resident)
__global__ void k_fps(const float* __restrict__ xyz, int* __restrict__ out, int n, int npoint) {
  extern __shared__ float sm[];
  float* sx = sm; float* sy = sm + n; float* sz = sm + 2 * n; float* sd = sm + 3 * n;
  __shared__ float rbv[32]; __shared__ int rbi[32]; __shared__ int s_far;
  int b = blockIdx.x; int t = threadIdx.x; int nt = blockDim.x;
  int lane = t & 31, wid = t >> 5, nw = nt >> 5;
  const float* p = xyz + (size_t)b * n * 3;
  for (int i = t; i < n; i += nt) { sx[i] = p[i*3]; sy[i] = p[i*3+1]; sz[i] = p[i*3+2]; sd[i] = 1e10f; }
  if (t == 0) s_far = 0;
  __syncthreads();
  for (int it = 0; it < npoint; ++it) {
    int far = s_far;
    if (t == 0) out[b * npoint + it] = far;
    float cx = sx[far], cy = sy[far], cz = sz[far];
    float best = -1.0f; int bi = 0;
    for (int i = t; i < n; i += nt) {
      float dx = sx[i]-cx, dy = sy[i]-cy, dz = sz[i]-cz;
      float d  = dx*dx + dy*dy + dz*dz;
      float nd = fminf(sd[i], d); sd[i] = nd;
      if (nd > best) { best = nd; bi = i; }
    }
    for (int off = 16; off > 0; off >>= 1) {
      float ov = __shfl_down(best, off);
      int   oi = __shfl_down(bi,   off);
      if (ov > best) { best = ov; bi = oi; }
    }
    if (lane == 0) { rbv[wid] = best; rbi[wid] = bi; }
    __syncthreads();
    if (t == 0) {
      float bb = rbv[0]; int ii = rbi[0];
      for (int w = 1; w < nw; ++w) if (rbv[w] > bb) { bb = rbv[w]; ii = rbi[w]; }
      s_far = ii;
    }
    __syncthreads();
  }
}

// ---------------------------------------------------------------- centroid gather
__global__ void k_gather_centroids(const float* __restrict__ xyz, const int* __restrict__ idx,
                                   float* __restrict__ out, int n, int S) {
  int tid = blockIdx.x * blockDim.x + threadIdx.x;
  int total = BATCH * S * 3;
  if (tid >= total) return;
  int c = tid % 3; int s = (tid / 3) % S; int b = tid / (3 * S);
  int j = idx[b * S + s];
  out[((size_t)b * S + s) * 3 + c] = xyz[((size_t)b * n + j) * 3 + c];
}

// ---------------------------------------------------------------- ball query (wave32 ballot)
__global__ void k_ball_query(const float* __restrict__ xyz, const float* __restrict__ cent,
                             int* __restrict__ out, int n, int S, int K, float r2) {
  int gwave = (int)((blockIdx.x * (size_t)blockDim.x + threadIdx.x) >> 5);
  int lane  = threadIdx.x & 31;
  if (gwave >= BATCH * S) return;          // wave-uniform
  int b = gwave / S;
  const float* c3 = cent + (size_t)gwave * 3;
  float cx = c3[0], cy = c3[1], cz = c3[2];
  const float* p = xyz + (size_t)b * n * 3;
  int* o = out + (size_t)gwave * K;
  int cnt = 0, first = 0; bool have = false;
  for (int base = 0; base < n && cnt < K; base += 32) {
    int j = base + lane;
    bool inb = false;
    if (j < n) {
      float dx = p[j*3]-cx, dy = p[j*3+1]-cy, dz = p[j*3+2]-cz;
      inb = (dx*dx + dy*dy + dz*dz) <= r2;
    }
    unsigned long long m = __ballot(inb) & 0xffffffffull;
    while (m && cnt < K) {
      int bit = __ffsll(m) - 1;
      m &= m - 1;
      if (!have) { first = base + bit; have = true; }
      if (lane == 0) o[cnt] = base + bit;
      ++cnt;
    }
  }
  for (int i = cnt + lane; i < K; i += 32) o[i] = first;
}

// ---------------------------------------------------------------- grouping (center + concat feats)
__global__ void k_group(const float* __restrict__ xyz, const float* __restrict__ pts,
                        const float* __restrict__ cent, const int* __restrict__ gi,
                        float* __restrict__ out, int n, int S, int K, int C) {
  int CT = 3 + C;
  long total = (long)BATCH * S * K * CT;
  long tid = blockIdx.x * (long)blockDim.x + threadIdx.x;
  if (tid >= total) return;
  int c = (int)(tid % CT); long g = tid / CT;
  long ss = g / K; int s = (int)(ss % S); int b = (int)(ss / S);
  int j = gi[g];
  float v;
  if (c < 3) v = xyz[((size_t)b * n + j) * 3 + c] - cent[((size_t)b * S + s) * 3 + c];
  else       v = pts[((size_t)b * n + j) * C + (c - 3)];
  out[tid] = v;
}

__global__ void k_group_all(const float* __restrict__ xyz, const float* __restrict__ pts,
                            float* __restrict__ out, int S, int C) {
  int CT = 3 + C;
  long total = (long)BATCH * S * CT;
  long tid = blockIdx.x * (long)blockDim.x + threadIdx.x;
  if (tid >= total) return;
  int c = (int)(tid % CT); long row = tid / CT;   // row = b*S + k
  out[tid] = (c < 3) ? xyz[(size_t)row * 3 + c] : pts[(size_t)row * C + (c - 3)];
}

// ---------------------------------------------------------------- operand packing for WMMA
// A: f16 row-major, K padded to Kp (mult of 32), zero-filled.
__global__ void k_pack_a(const float* __restrict__ X, _Float16* __restrict__ Xp,
                         int M, int K, int Kp) {
  long total = (long)M * Kp;
  long tid = blockIdx.x * (long)blockDim.x + threadIdx.x;
  if (tid >= total) return;
  int k = (int)(tid % Kp); long row = tid / Kp;
  Xp[tid] = (_Float16)((k < K) ? X[(size_t)row * K + k] : 0.0f);
}

// W: packed per (n-tile, k-chunk, lane, 16 halves) = exact B fragment each lane loads.
// element e of lane's fragment corresponds to kb = kci*32 + (lane>=16?8:0) + (e<8 ? e : 16+e-8).
__global__ void k_pack_w(const float* __restrict__ W, _Float16* __restrict__ Wp,
                         int K, int N, int Kp) {
  int nK = Kp >> 5;
  int ntiles = (N + 15) >> 4;
  long total = (long)ntiles * nK * 512;
  long tid = blockIdx.x * (long)blockDim.x + threadIdx.x;
  if (tid >= total) return;
  int e    = (int)(tid & 15);
  int lane = (int)((tid >> 4) & 31);
  long grp = tid >> 9;                    // nt*nK + kci
  int kci  = (int)(grp % nK);
  int nt   = (int)(grp / nK);
  int kb   = kci * 32 + ((lane >= 16) ? 8 : 0) + ((e < 8) ? e : 16 + (e - 8));
  int col  = nt * 16 + (lane & 15);
  float v  = (kb < K && col < N) ? W[(size_t)kb * N + col] : 0.0f;
  Wp[tid] = (_Float16)v;
}

// ---------------------------------------------------------------- WMMA GEMM: Y = Xp*W + bias
// One wave per 16(M) x 64(N) tile: 4 accumulators, 4 wmma per 32-wide K chunk.
// Requires: M % 16 == 0, ((N+15)/16) % 4 == 0 (true for all layers here).
#define WMMA_F16(Av, Bv, Cc) __builtin_amdgcn_wmma_f32_16x16x32_f16(false, (Av), false, (Bv), (short)0, (Cc), false, false)

__global__ void k_gemm_wmma4(const _Float16* __restrict__ Xp, const _Float16* __restrict__ Wp,
                             const float* __restrict__ bias, float* __restrict__ Y,
                             int M, int N, int Kp) {
  int ntiles  = (N + 15) >> 4;
  int ngroups = ntiles >> 2;
  int wave = (int)((blockIdx.x * (size_t)blockDim.x + threadIdx.x) >> 5);
  int lane = threadIdx.x & 31;
  int mt = wave / ngroups, ng = wave % ngroups;
  if (mt * 16 >= M) return;                     // wave-uniform -> EXEC all-ones at the WMMA
  int lm = lane & 15;
  bool hi = lane >= 16;
  int m  = mt * 16 + lm;
  int nK = Kp >> 5;
  const _Float16* arow  = Xp + (size_t)m * Kp + (hi ? 8 : 0);
  size_t tstride = (size_t)nK * 512;            // halves per packed n-tile
  const _Float16* bbase = Wp + (size_t)(ng * 4) * tstride + (size_t)lane * 16;
  v8f acc0 = {}, acc1 = {}, acc2 = {}, acc3 = {};
  for (int kci = 0; kci < nK; ++kci) {
    union { v16h v; v8h h[2]; } Af, B0, B1, B2, B3;
    const _Float16* a = arow + (size_t)kci * 32;
    Af.h[0] = *(const v8h*)(a);
    Af.h[1] = *(const v8h*)(a + 16);
    const _Float16* b = bbase + (size_t)kci * 512;
    B0.h[0] = *(const v8h*)(b);
    B0.h[1] = *(const v8h*)(b + 8);
    B1.h[0] = *(const v8h*)(b + tstride);
    B1.h[1] = *(const v8h*)(b + tstride + 8);
    B2.h[0] = *(const v8h*)(b + 2 * tstride);
    B2.h[1] = *(const v8h*)(b + 2 * tstride + 8);
    B3.h[0] = *(const v8h*)(b + 3 * tstride);
    B3.h[1] = *(const v8h*)(b + 3 * tstride + 8);
    if (kci + 1 < nK) __builtin_prefetch(arow + (size_t)(kci + 1) * 32, 0, 1);  // uniform branch
    acc0 = WMMA_F16(Af.v, B0.v, acc0);
    acc1 = WMMA_F16(Af.v, B1.v, acc1);
    acc2 = WMMA_F16(Af.v, B2.v, acc2);
    acc3 = WMMA_F16(Af.v, B3.v, acc3);
  }
  int rbase = mt * 16 + (hi ? 8 : 0);
  union { v8f v; float f[8]; } Cv;
#pragma unroll
  for (int t = 0; t < 4; ++t) {
    int ncol = (ng * 4 + t) * 16 + lm;
    if (ncol >= N) continue;                    // only ragged for N=50
    Cv.v = (t == 0) ? acc0 : (t == 1) ? acc1 : (t == 2) ? acc2 : acc3;
    float bb = bias ? bias[ncol] : 0.0f;
#pragma unroll
    for (int r = 0; r < 8; ++r)
      Y[(size_t)(rbase + r) * N + ncol] = Cv.f[r] + bb;
  }
}

// ---------------------------------------------------------------- BatchNorm stats + norm/ReLU
__global__ void k_bn_stats(const float* __restrict__ Y, float* __restrict__ mean,
                           float* __restrict__ var, int M, int C) {
  __shared__ float ss[256], sq[256];
  int c = blockIdx.x;
  float s = 0.f, q = 0.f;
  for (int i = threadIdx.x; i < M; i += blockDim.x) {
    float v = Y[(size_t)i * C + c]; s += v; q += v * v;
  }
  ss[threadIdx.x] = s; sq[threadIdx.x] = q;
  __syncthreads();
  for (int o = blockDim.x >> 1; o > 0; o >>= 1) {
    if (threadIdx.x < o) { ss[threadIdx.x] += ss[threadIdx.x + o]; sq[threadIdx.x] += sq[threadIdx.x + o]; }
    __syncthreads();
  }
  if (threadIdx.x == 0) {
    float mu = ss[0] / (float)M;
    mean[c] = mu;
    var[c]  = sq[0] / (float)M - mu * mu;
  }
}

__global__ void k_bn_relu(float* __restrict__ Y, const float* __restrict__ mean,
                          const float* __restrict__ var, const float* __restrict__ g,
                          const float* __restrict__ be, int M, int C) {
  size_t tid = blockIdx.x * (size_t)blockDim.x + threadIdx.x;
  size_t total = (size_t)M * C;
  if (tid >= total) return;
  int c = (int)(tid % C);
  float v = (Y[tid] - mean[c]) * rsqrtf(var[c] + 1e-5f) * g[c] + be[c];
  Y[tid] = fmaxf(v, 0.0f);
}

// ---------------------------------------------------------------- max over group axis
__global__ void k_maxpool(const float* __restrict__ in, float* __restrict__ out,
                          int BS, int K, int C) {
  int tid = blockIdx.x * blockDim.x + threadIdx.x;
  int total = BS * C;
  if (tid >= total) return;
  int c = tid % C; int bs = tid / C;
  float m = -3.4e38f;
  for (int k = 0; k < K; ++k) m = fmaxf(m, in[((size_t)bs * K + k) * C + c]);
  out[(size_t)bs * C + c] = m;
}

// ---------------------------------------------------------------- 3-NN (idx + normalized 1/d weights)
__global__ void k_knn3(const float* __restrict__ q, const float* __restrict__ ref,
                       int* __restrict__ oidx, float* __restrict__ ow, int Nq, int S) {
  int tid = blockIdx.x * blockDim.x + threadIdx.x;
  int total = BATCH * Nq;
  if (tid >= total) return;
  int b = tid / Nq;
  const float* qp = q + (size_t)tid * 3;
  float qx = qp[0], qy = qp[1], qz = qp[2];
  const float* rp = ref + (size_t)b * S * 3;
  float d0 = 3e38f, d1 = 3e38f, d2 = 3e38f; int i0 = 0, i1 = 0, i2 = 0;
  for (int j = 0; j < S; ++j) {
    float dx = rp[j*3]-qx, dy = rp[j*3+1]-qy, dz = rp[j*3+2]-qz;
    float d = dx*dx + dy*dy + dz*dz;
    if      (d < d0) { d2=d1;i2=i1; d1=d0;i1=i0; d0=d;i0=j; }
    else if (d < d1) { d2=d1;i2=i1; d1=d;i1=j; }
    else if (d < d2) { d2=d;i2=j; }
  }
  float w0 = 1.f/(d0+1e-8f), w1 = 1.f/(d1+1e-8f), w2 = 1.f/(d2+1e-8f);
  float s = w0 + w1 + w2;
  oidx[tid*3] = i0; oidx[tid*3+1] = i1; oidx[tid*3+2] = i2;
  ow[tid*3] = w0/s; ow[tid*3+1] = w1/s; ow[tid*3+2] = w2/s;
}

// ---------------------------------------------------------------- FP input builders
__global__ void k_fp_concat_interp(const float* __restrict__ pts1, int C1,
                                   const float* __restrict__ pts2, int C2,
                                   const int* __restrict__ kidx, const float* __restrict__ kw,
                                   float* __restrict__ out, int Nq, int Sref) {
  int CT = C1 + C2;
  long total = (long)BATCH * Nq * CT;
  long tid = blockIdx.x * (long)blockDim.x + threadIdx.x;
  if (tid >= total) return;
  int c = (int)(tid % CT); long row = tid / CT; int b = (int)(row / Nq);
  float v;
  if (c < C1) v = pts1[(size_t)row * C1 + c];
  else {
    int c2 = c - C1;
    const int* ki = kidx + row * 3; const float* wv = kw + row * 3;
    v = wv[0] * pts2[((size_t)b * Sref + ki[0]) * C2 + c2]
      + wv[1] * pts2[((size_t)b * Sref + ki[1]) * C2 + c2]
      + wv[2] * pts2[((size_t)b * Sref + ki[2]) * C2 + c2];
  }
  out[tid] = v;
}

__global__ void k_fp1_input(const float* __restrict__ cls, const float* __restrict__ xyz0,
                            const float* __restrict__ pts0, const float* __restrict__ pts2,
                            const int* __restrict__ kidx, const float* __restrict__ kw,
                            float* __restrict__ out, int Nq, int Sref, int C2) {
  int CT = 22 + C2;  // 16 one-hot + 3 xyz + 3 extra + 128 interp = 150
  long total = (long)BATCH * Nq * CT;
  long tid = blockIdx.x * (long)blockDim.x + threadIdx.x;
  if (tid >= total) return;
  int c = (int)(tid % CT); long row = tid / CT; int b = (int)(row / Nq);
  float v;
  if      (c < 16) v = cls[b * 16 + c];
  else if (c < 19) v = xyz0[(size_t)row * 3 + (c - 16)];
  else if (c < 22) v = pts0[(size_t)row * 3 + (c - 19)];
  else {
    int c2 = c - 22;
    const int* ki = kidx + row * 3; const float* wv = kw + row * 3;
    v = wv[0] * pts2[((size_t)b * Sref + ki[0]) * C2 + c2]
      + wv[1] * pts2[((size_t)b * Sref + ki[1]) * C2 + c2]
      + wv[2] * pts2[((size_t)b * Sref + ki[2]) * C2 + c2];
  }
  out[tid] = v;
}

__global__ void k_fp3_input(const float* __restrict__ p1, const float* __restrict__ p2,
                            float* __restrict__ out, int Nq, int C1, int C2) {
  int CT = C1 + C2;
  long total = (long)BATCH * Nq * CT;
  long tid = blockIdx.x * (long)blockDim.x + threadIdx.x;
  if (tid >= total) return;
  int c = (int)(tid % CT); long row = tid / CT; int b = (int)(row / Nq);
  out[tid] = (c < C1) ? p1[(size_t)row * C1 + c] : p2[(size_t)b * C2 + (c - C1)];
}

// ---------------------------------------------------------------- log-softmax
__global__ void k_log_softmax(const float* __restrict__ in, float* __restrict__ out, int M, int C) {
  int tid = blockIdx.x * blockDim.x + threadIdx.x;
  if (tid >= M) return;
  const float* x = in + (size_t)tid * C;
  float mx = -3.4e38f;
  for (int c = 0; c < C; ++c) mx = fmaxf(mx, x[c]);
  float s = 0.f;
  for (int c = 0; c < C; ++c) s += expf(x[c] - mx);
  float l = logf(s);
  float* o = out + (size_t)tid * C;
  for (int c = 0; c < C; ++c) o[c] = x[c] - mx - l;
}

// ================================================================ host
static inline unsigned cdiv_u(long a, long b) { return (unsigned)((a + b - 1) / b); }

extern "C" void kernel_launch(void* const* d_in, const int* in_sizes, int n_in,
                              void* d_out, int out_size, void* d_ws, size_t ws_size,
                              hipStream_t stream) {
  (void)in_sizes; (void)out_size; (void)ws_size;
  const float* IN  = (const float*)d_in[0];
  const float* CLS = (const float*)d_in[1];
  const float* P[72] = {nullptr};
  for (int i = 0; i < 72 && i < n_in; ++i) P[i] = (const float*)d_in[i];

  // workspace layout (float units); total ~83M floats (~335 MB)
  float* Wb = (float*)d_ws;
  size_t o = 0;
  auto A = [&](size_t cnt) -> float* { float* p = Wb + o; o += cnt; return p; };
  float* l0_xyz = A((size_t)BATCH * NPTS * 3);
  float* l0_pts = A((size_t)BATCH * NPTS * 3);
  float* l1_xyz = A((size_t)BATCH * 512 * 3);
  float* l1_pts = A((size_t)BATCH * 512 * 128);   // SA1 output
  float* l1_fp  = A((size_t)BATCH * 512 * 128);   // FP2 output
  float* l2_xyz = A((size_t)BATCH * 128 * 3);
  float* l2_sa  = A((size_t)BATCH * 128 * 256);   // SA2 output
  float* l2_fp  = A((size_t)BATCH * 128 * 256);   // FP3 output
  float* l3_pts = A((size_t)BATCH * 1024);        // SA3 output
  int*   fps1   = (int*)A((size_t)BATCH * 512);
  int*   fps2   = (int*)A((size_t)BATCH * 128);
  int*   ball1  = (int*)A((size_t)BATCH * 512 * 32);
  int*   ball2  = (int*)A((size_t)BATCH * 128 * 64);
  int*   kidx   = (int*)A((size_t)BATCH * NPTS * 3);
  float* kw     = A((size_t)BATCH * NPTS * 3);
  float* meanb  = A(1024);
  float* varb   = A(1024);
  _Float16* aPack = (_Float16*)A((size_t)10600000);   // >= 21.0M halves (max M*Kp = 131072*160)
  _Float16* wPack = (_Float16*)A((size_t)300000);     // >= 0.525M halves (max Kp*Np = 512*1024)
  float* pb0    = A((size_t)33554432);                // 262144 x 128 max
  float* pb1    = A((size_t)33554432);

  const int T = 256;
  // GEMM with pre-packed operands (M % 16 == 0 for every layer here)
  auto gemm = [&](const float* X, const float* W, const float* bias, float* Y,
                  int M, int K, int N) {
    int Kp = (K + 31) & ~31;
    int ntiles = (N + 15) >> 4;             // always a multiple of 4 here
    int nK = Kp >> 5;
    k_pack_a<<<cdiv_u((long)M * Kp, T), T, 0, stream>>>(X, aPack, M, K, Kp);
    k_pack_w<<<cdiv_u((long)ntiles * nK * 512, T), T, 0, stream>>>(W, wPack, K, N, Kp);
    long waves = (long)(M / 16) * (ntiles / 4);
    k_gemm_wmma4<<<cdiv_u(waves * 32, T), T, 0, stream>>>(aPack, wPack, bias, Y, M, N, Kp);
  };
  auto layer = [&](const float* X, float* Y, int wi, int M, int K, int N) {
    gemm(X, P[wi], P[wi + 1], Y, M, K, N);
    k_bn_stats<<<N, 256, 0, stream>>>(Y, meanb, varb, M, N);
    k_bn_relu<<<cdiv_u((long)M * N, T), T, 0, stream>>>(Y, meanb, varb, P[wi + 2], P[wi + 3], M, N);
  };

  // stage 0: (B,6,N) -> l0_xyz (B,N,3), l0_pts (B,N,3)
  k_split_input<<<cdiv_u((long)BATCH * NPTS * 6, T), T, 0, stream>>>(IN, l0_xyz, l0_pts, NPTS);

  // ---------------- SA1: 4096 -> 512 centroids, K=32, MLP 6->64->64->128
  k_fps<<<BATCH, 512, (size_t)16 * NPTS, stream>>>(l0_xyz, fps1, NPTS, 512);
  k_gather_centroids<<<cdiv_u((long)BATCH * 512 * 3, T), T, 0, stream>>>(l0_xyz, fps1, l1_xyz, NPTS, 512);
  k_ball_query<<<cdiv_u((long)BATCH * 512 * 32, T), T, 0, stream>>>(l0_xyz, l1_xyz, ball1, NPTS, 512, 32, 0.04f);
  k_group<<<cdiv_u((long)BATCH * 512 * 32 * 6, T), T, 0, stream>>>(l0_xyz, l0_pts, l1_xyz, ball1, pb0, NPTS, 512, 32, 3);
  const int M1 = BATCH * 512 * 32;
  layer(pb0, pb1, 2,  M1, 6,  64);
  layer(pb1, pb0, 6,  M1, 64, 64);
  layer(pb0, pb1, 10, M1, 64, 128);
  k_maxpool<<<cdiv_u((long)BATCH * 512 * 128, T), T, 0, stream>>>(pb1, l1_pts, BATCH * 512, 32, 128);

  // ---------------- SA2: 512 -> 128 centroids, K=64, MLP 131->128->128->256
  k_fps<<<BATCH, 512, (size_t)16 * 512, stream>>>(l1_xyz, fps2, 512, 128);
  k_gather_centroids<<<cdiv_u((long)BATCH * 128 * 3, T), T, 0, stream>>>(l1_xyz, fps2, l2_xyz, 512, 128);
  k_ball_query<<<cdiv_u((long)BATCH * 128 * 32, T), T, 0, stream>>>(l1_xyz, l2_xyz, ball2, 512, 128, 64, 0.16f);
  k_group<<<cdiv_u((long)BATCH * 128 * 64 * 131, T), T, 0, stream>>>(l1_xyz, l1_pts, l2_xyz, ball2, pb0, 512, 128, 64, 128);
  const int M2 = BATCH * 128 * 64;
  layer(pb0, pb1, 14, M2, 131, 128);
  layer(pb1, pb0, 18, M2, 128, 128);
  layer(pb0, pb1, 22, M2, 128, 256);
  k_maxpool<<<cdiv_u((long)BATCH * 128 * 256, T), T, 0, stream>>>(pb1, l2_sa, BATCH * 128, 64, 256);

  // ---------------- SA3 (group_all): MLP 259->256->512->1024, max over 128
  k_group_all<<<cdiv_u((long)BATCH * 128 * 259, T), T, 0, stream>>>(l2_xyz, l2_sa, pb0, 128, 256);
  const int M3 = BATCH * 128;
  layer(pb0, pb1, 26, M3, 259, 256);
  layer(pb1, pb0, 30, M3, 256, 512);
  layer(pb0, pb1, 34, M3, 512, 1024);
  k_maxpool<<<cdiv_u((long)BATCH * 1024, T), T, 0, stream>>>(pb1, l3_pts, BATCH, 128, 1024);

  // ---------------- FP3: broadcast l3 + concat l2_sa -> 1280 -> 256 -> 256
  k_fp3_input<<<cdiv_u((long)BATCH * 128 * 1280, T), T, 0, stream>>>(l2_sa, l3_pts, pb0, 128, 256, 1024);
  layer(pb0, pb1,  38, M3, 1280, 256);
  layer(pb1, l2_fp, 42, M3, 256, 256);

  // ---------------- FP2: 3NN(l1<-l2), concat(l1_pts,interp)=384 -> 256 -> 128
  k_knn3<<<cdiv_u((long)BATCH * 512, T), T, 0, stream>>>(l1_xyz, l2_xyz, kidx, kw, 512, 128);
  k_fp_concat_interp<<<cdiv_u((long)BATCH * 512 * 384, T), T, 0, stream>>>(l1_pts, 128, l2_fp, 256, kidx, kw, pb0, 512, 128);
  const int M4 = BATCH * 512;
  layer(pb0, pb1,  46, M4, 384, 256);
  layer(pb1, l1_fp, 50, M4, 256, 128);

  // ---------------- FP1: 3NN(l0<-l1), concat(cls16+xyz3+pts3, interp128)=150 -> 128x3
  k_knn3<<<cdiv_u((long)BATCH * NPTS, T), T, 0, stream>>>(l0_xyz, l1_xyz, kidx, kw, NPTS, 512);
  k_fp1_input<<<cdiv_u((long)BATCH * NPTS * 150, T), T, 0, stream>>>(CLS, l0_xyz, l0_pts, l1_fp, kidx, kw, pb0, NPTS, 512, 128);
  const int M5 = BATCH * NPTS;
  layer(pb0, pb1, 54, M5, 150, 128);
  layer(pb1, pb0, 58, M5, 128, 128);
  layer(pb0, pb1, 62, M5, 128, 128);

  // ---------------- head: 128 -> 128 (bn+relu) -> 50 (+bias) -> log_softmax
  layer(pb1, pb0, 66, M5, 128, 128);
  gemm(pb0, P[70], P[71], pb1, M5, 128, NCLS);
  k_log_softmax<<<cdiv_u((long)M5, T), T, 0, stream>>>(pb1, (float*)d_out, M5, NCLS);
}
